// TopkSingleSelfAttention_73701638799896
// MI455X (gfx1250) — matrix operand order repaired and checked by reading, not compiled, not run on hardware
//
#include <hip/hip_runtime.h>
#include <hip/hip_bf16.h>
#include <math.h>

// ---------------------------------------------------------------------------
// CDNA5 (gfx1250) wave32 WMMA types
// ---------------------------------------------------------------------------
typedef __attribute__((ext_vector_type(16))) __bf16 v16bf;
typedef __attribute__((ext_vector_type(8)))  __bf16 v8bf;
typedef __attribute__((ext_vector_type(4)))  __bf16 v4bf;
typedef __attribute__((ext_vector_type(8)))  float  v8f;

// Branch-free tanh-GELU: tanh(u) = copysign(1 - 2/(exp(2|u|)+1), u)
__device__ __forceinline__ float gelu_tanh(float x) {
  const float k0 = 0.7978845608028654f;  // sqrt(2/pi)
  const float k1 = 0.044715f;
  float u  = k0 * (x + k1 * x * x * x);
  float e  = __expf(2.0f * fabsf(u));                       // v_exp_f32
  float t  = 1.0f - 2.0f * __builtin_amdgcn_rcpf(e + 1.0f); // v_rcp_f32
  float th = copysignf(t, u);
  return 0.5f * x * (1.0f + th);
}

__device__ __forceinline__ void load16(const float* p, float* out) {
  const float4* q = (const float4*)p;
  float4 t0 = q[0], t1 = q[1], t2 = q[2], t3 = q[3];
  const float* s0 = (const float*)&t0; const float* s1 = (const float*)&t1;
  const float* s2 = (const float*)&t2; const float* s3 = (const float*)&t3;
#pragma unroll
  for (int i = 0; i < 4; ++i) {
    out[i] = s0[i]; out[4 + i] = s1[i]; out[8 + i] = s2[i]; out[12 + i] = s3[i];
  }
}
__device__ __forceinline__ void load16(const __bf16* p, float* out) {
  const v8bf* q = (const v8bf*)p;
  v8bf t0 = q[0], t1 = q[1];
#pragma unroll
  for (int i = 0; i < 8; ++i) { out[i] = (float)t0[i]; out[8 + i] = (float)t1[i]; }
}

// ---------------------------------------------------------------------------
// Tiled WMMA GEMM:  C[M,N] = gelu(A[M,K] @ W[K,N] + bias[N])  (bf16 out)
// Block tile 128x128, 8 waves (2x4), each wave 64x32 = 4x2 WMMA tiles.
// Double-buffered LDS, 1 barrier / K-step. B staged TRANSPOSED (register
// path, fp32->bf16 convert). A staged:
//   - AT=float : register path with convert (GEMM1, A = x fp32)
//   - AT=bf16  : GLOBAL_LOAD_ASYNC_TO_LDS_B128 (GEMM2, A = h1 bf16),
//                synced with s_wait_asynccnt + workgroup barrier.
// ---------------------------------------------------------------------------
#define BM 128
#define BN 128
#define BK 32
#define LDP (BK + 8)   // padded pitch: 20-dword lane stride -> conflict-free

template <typename AT>
__global__ __launch_bounds__(256) void gemm_bias_gelu_bf16(
    const AT* __restrict__ A, const float* __restrict__ W,
    const float* __restrict__ bias, __bf16* __restrict__ C,
    int M, int N, int K) {
  constexpr bool ASYNC_A = (sizeof(AT) == 2);   // bf16 A -> async DMA staging

  __shared__ __bf16 As[2][BM][LDP];   // row-major tile
  __shared__ __bf16 Bs[2][BN][LDP];   // TRANSPOSED tile: [col][k]

  const int tid  = threadIdx.x;
  const int wave = tid >> 5;
  const int lane = tid & 31;
  const int half = lane >> 4;
  const int l16  = lane & 15;
  const int wm   = wave >> 2;          // 0..1 (64 rows)
  const int wn   = wave & 3;           // 0..3 (32 cols)
  const int bm   = blockIdx.y * BM;
  const int bn   = blockIdx.x * BN;

  const int ra = tid >> 1, ca = (tid & 1) * 16;   // A: 16 contiguous elems
  const int cb = (lane & 31) * 4;                 // B: 4x4 sub-block
  const int rb = (tid >> 5) * 4;

  float  aPre[16];
  float4 wPre[4];

  // ---- A staging -----------------------------------------------------------
  auto fetchA = [&](int k0, int buf) {
    if constexpr (ASYNC_A) {
      // Per-lane async copy of 32 B: LDS[vdst+off] = MEM[saddr+vaddr+off]
      unsigned ldsa = (unsigned)(size_t)&As[buf][ra][ca];
      unsigned goff = (unsigned)(((size_t)(bm + ra) * K + (k0 + ca)) * 2u);
      asm volatile(
          "global_load_async_to_lds_b128 %0, %1, %2\n\t"
          "global_load_async_to_lds_b128 %0, %1, %2 offset:16"
          :: "v"(ldsa), "v"(goff), "s"((unsigned long long)(size_t)A)
          : "memory");
    } else {
      load16(&A[(size_t)(bm + ra) * K + (k0 + ca)], aPre);
    }
  };
  auto commitA = [&](int buf) {
    if constexpr (!ASYNC_A) {
      v8bf t0, t1;
#pragma unroll
      for (int i = 0; i < 8; ++i) { t0[i] = (__bf16)aPre[i]; t1[i] = (__bf16)aPre[8 + i]; }
      *(v8bf*)&As[buf][ra][ca]     = t0;
      *(v8bf*)&As[buf][ra][ca + 8] = t1;
    }
  };
  auto waitA = [&]() {
    if constexpr (ASYNC_A)
      asm volatile("s_wait_asynccnt 0x0" ::: "memory");
  };

  // ---- B staging (register path, transpose + convert) ----------------------
  auto fetchB = [&](int k0) {
#pragma unroll
    for (int i = 0; i < 4; ++i)
      wPre[i] = *(const float4*)&W[(size_t)(k0 + rb + i) * N + (bn + cb)];
  };
  auto commitB = [&](int buf) {
#pragma unroll
    for (int cc = 0; cc < 4; ++cc) {
      v4bf t;
#pragma unroll
      for (int i = 0; i < 4; ++i) t[i] = (__bf16)((const float*)&wPre[i])[cc];
      *(v4bf*)&Bs[buf][cb + cc][rb] = t;   // transposed store
    }
  };

  v8f acc[4][2] = {};

  fetchA(0, 0);
  fetchB(0);
  commitA(0);
  commitB(0);
  waitA();
  __syncthreads();

  const int NK = K / BK;
  for (int t = 0; t < NK; ++t) {
    const int buf = t & 1;
    if (t + 1 < NK) {
      fetchA((t + 1) * BK, buf ^ 1);
      fetchB((t + 1) * BK);
      if (t + 2 < NK) {  // global_prefetch_b8 two tiles ahead
        __builtin_prefetch(&A[(size_t)(bm + ra) * K + ((t + 2) * BK + ca)], 0, 1);
        __builtin_prefetch(&W[(size_t)((t + 2) * BK + rb) * N + (bn + cb)], 0, 1);
      }
    }

    // Fragments: contiguous 16B LDS loads (ds_load_b128)
    v16bf aF[4], bF[2];
#pragma unroll
    for (int mt = 0; mt < 4; ++mt) {
      int row = wm * 64 + mt * 16 + l16;
      v8bf lo = *(const v8bf*)&As[buf][row][half * 8];        // K = h*8..h*8+7
      v8bf hi = *(const v8bf*)&As[buf][row][16 + half * 8];   // K = 16+h*8..
      aF[mt] = __builtin_shufflevector(lo, hi, 0, 1, 2, 3, 4, 5, 6, 7,
                                       8, 9, 10, 11, 12, 13, 14, 15);
    }
#pragma unroll
    for (int nt = 0; nt < 2; ++nt) {
      int col = wn * 32 + nt * 16 + l16;
      v8bf lo = *(const v8bf*)&Bs[buf][col][half * 8];        // K = h*8..h*8+7
      v8bf hi = *(const v8bf*)&Bs[buf][col][16 + half * 8];
      bF[nt] = __builtin_shufflevector(lo, hi, 0, 1, 2, 3, 4, 5, 6, 7,
                                       8, 9, 10, 11, 12, 13, 14, 15);
    }

#pragma unroll
    for (int mt = 0; mt < 4; ++mt)
#pragma unroll
      for (int nt = 0; nt < 2; ++nt)
        acc[mt][nt] = __builtin_amdgcn_wmma_f32_16x16x32_bf16(
            false, aF[mt], false, bF[nt], (short)0, acc[mt][nt], false, false);

    if (t + 1 < NK) {
      commitA(buf ^ 1);
      commitB(buf ^ 1);
      waitA();          // next A tile resident before the barrier releases
    }
    __syncthreads();
  }

  // Epilogue: bias + GELU, store bf16. C/D layout: row = vgpr + half*8.
#pragma unroll
  for (int mt = 0; mt < 4; ++mt) {
#pragma unroll
    for (int nt = 0; nt < 2; ++nt) {
      int col = bn + wn * 32 + nt * 16 + l16;
      float bv = bias[col];
#pragma unroll
      for (int r = 0; r < 8; ++r) {
        int row = bm + wm * 64 + mt * 16 + half * 8 + r;
        C[(size_t)row * N + col] = (__bf16)gelu_tanh(acc[mt][nt][r] + bv);
      }
    }
  }
}

// ---------------------------------------------------------------------------
// alpha[row] = dot(h2[row,:], Wa) + ba  (one wave32 per row)
// ---------------------------------------------------------------------------
__global__ __launch_bounds__(256) void alpha_dot(
    const __bf16* __restrict__ h2, const float* __restrict__ Wa,
    const float* __restrict__ ba, float* __restrict__ alpha,
    int rows, int D2) {
  int row  = blockIdx.x * 8 + (threadIdx.x >> 5);
  int lane = threadIdx.x & 31;
  if (row >= rows) return;
  const v8bf* p = (const v8bf*)(h2 + (size_t)row * D2 + lane * 16);
  v8bf h0 = p[0], h1v = p[1];
  float wv[16];
  load16(Wa + lane * 16, wv);
  float s = 0.0f;
#pragma unroll
  for (int i = 0; i < 8; ++i) s += (float)h0[i] * wv[i] + (float)h1v[i] * wv[8 + i];
#pragma unroll
  for (int off = 16; off > 0; off >>= 1) s += __shfl_down(s, off, 32);
  if (lane == 0) alpha[row] = s + ba[0];
}

// ---------------------------------------------------------------------------
// Top-64 per batch (softmax skipped: monotonic => identical indices).
// ---------------------------------------------------------------------------
__global__ __launch_bounds__(256) void topk_kernel(
    const float* __restrict__ alpha, int* __restrict__ indices,
    int S, int KT) {
  __shared__ float vals[2048];
  __shared__ float rv[256];
  __shared__ int   ri[256];
  const int b = blockIdx.x, tid = threadIdx.x;
  for (int i = tid; i < S; i += 256) vals[i] = alpha[(size_t)b * S + i];
  __syncthreads();
  for (int it = 0; it < KT; ++it) {
    float best = -3.4e38f; int bi = 0;
    for (int i = tid; i < S; i += 256) {
      float v = vals[i];
      if (v > best) { best = v; bi = i; }
    }
    rv[tid] = best; ri[tid] = bi;
    __syncthreads();
    for (int s = 128; s > 0; s >>= 1) {
      if (tid < s) {
        bool take = (rv[tid + s] > rv[tid]) ||
                    (rv[tid + s] == rv[tid] && ri[tid + s] < ri[tid]);
        if (take) { rv[tid] = rv[tid + s]; ri[tid] = ri[tid + s]; }
      }
      __syncthreads();
    }
    if (tid == 0) { indices[b * KT + it] = ri[0]; vals[ri[0]] = -3.4e38f; }
    __syncthreads();
  }
}

// ---------------------------------------------------------------------------
// candidate[b,k,:] = x[b, idx[b,k], :]
// ---------------------------------------------------------------------------
__global__ __launch_bounds__(256) void gather_kernel(
    const float* __restrict__ x, const int* __restrict__ indices,
    float* __restrict__ out, int D, int S, int KT) {
  int b = blockIdx.y, k = blockIdx.x;
  int idx = indices[b * KT + k];
  const float4* src = (const float4*)(x + ((size_t)b * S + idx) * D);
  float4*       dst = (float4*)(out + ((size_t)b * KT + k) * D);
  dst[threadIdx.x] = src[threadIdx.x];
}

// ---------------------------------------------------------------------------
extern "C" void kernel_launch(void* const* d_in, const int* in_sizes, int n_in,
                              void* d_out, int out_size, void* d_ws, size_t ws_size,
                              hipStream_t stream) {
  const float* x  = (const float*)d_in[0];
  const float* W1 = (const float*)d_in[1];
  const float* b1 = (const float*)d_in[2];
  const float* W2 = (const float*)d_in[3];
  const float* b2 = (const float*)d_in[4];
  const float* Wa = (const float*)d_in[5];
  const float* ba = (const float*)d_in[6];

  const int B = 32, S = 2048, D = 1024, D1 = 1024, D2 = 512, KT = 64;
  const int M = B * S;  // 65536

  // Workspace layout (~192.3 MB)
  char* ws = (char*)d_ws;
  __bf16* h1    = (__bf16*)(ws);                                    // 128 MB
  __bf16* h2    = (__bf16*)(ws + (size_t)128 * 1024 * 1024);        //  64 MB
  float*  alpha = (float*)(ws + (size_t)(128 + 64) * 1024 * 1024);  // 256 KB

  int*   outIdx  = (int*)d_out;
  float* outCand = (float*)d_out + (size_t)B * KT;

  gemm_bias_gelu_bf16<float><<<dim3(D1 / BN, M / BM), 256, 0, stream>>>(
      x, W1, b1, h1, M, D1, D);
  gemm_bias_gelu_bf16<__bf16><<<dim3(D2 / BN, M / BM), 256, 0, stream>>>(
      h1, W2, b2, h2, M, D2, D1);
  alpha_dot<<<M / 8, 256, 0, stream>>>(h2, Wa, ba, alpha, M, D2);
  topk_kernel<<<B, 256, 0, stream>>>(alpha, outIdx, S, KT);
  gather_kernel<<<dim3(KT, B), 256, 0, stream>>>(x, outIdx, outCand, D, S, KT);
}
